// Sepconv_71871982731747
// MI455X (gfx1250) — matrix operand order repaired and checked by reading, not compiled, not run on hardware
//
#include <hip/hip_runtime.h>
#include <cstdint>
#include <cstddef>

// ---------------------------------------------------------------------------
// SepConv U-Net for MI455X (gfx1250).
//  * All 3x3 convs: implicit GEMM on v_wmma_f32_16x16x32_f16 (wave32).
//  * Activations: NHWC fp16, channels padded to multiple of 32 (padded lanes
//    are kept zero everywhere so they contribute nothing to any GEMM).
//  * Weights: packed on device into the exact per-lane WMMA B-fragment layout
//    so each lane's B operand is one contiguous 32B load.
//  * Conv kernel templated on NCHUNK (= Cinp/32) with the chunk loop fully
//    unrolled: straight-line WMMA runs with clustered loads, plus
//    global_prefetch of the next tap's weight fragments.
//  * Final 51x51 separable adaptive conv: LDS-staged kernels, factored
//    horizontal-then-vertical reduction, fp32.
// ---------------------------------------------------------------------------

typedef __attribute__((ext_vector_type(16))) _Float16 v16h;
typedef __attribute__((ext_vector_type(8)))  _Float16 v8h;
typedef __attribute__((ext_vector_type(8)))  float    v8f;

#define SEP_KS  51
#define SEP_PAD 25

__device__ __forceinline__ int iclampi(int v, int lo, int hi) {
  return v < lo ? lo : (v > hi ? hi : v);
}

// ---------------------------------------------------------------------------
// Pack the two fp32 NCHW inputs into one NHWC fp16 buffer [P][32], ch 0..5
// valid, 6..31 zero.
// ---------------------------------------------------------------------------
__global__ __launch_bounds__(256) void pack_input_kernel(
    const float* __restrict__ in1, const float* __restrict__ in2,
    _Float16* __restrict__ out, int P) {
  int i = blockIdx.x * 256 + threadIdx.x;
  if (i >= P * 32) return;
  int c = i & 31;
  int p = i >> 5;
  float v = 0.0f;
  if (c < 3)      v = in1[(size_t)c * P + p];
  else if (c < 6) v = in2[(size_t)(c - 3) * P + p];
  out[i] = (_Float16)v;
}

// ---------------------------------------------------------------------------
// Pack fp32 OIHW conv weights into per-lane WMMA B fragments (fp16).
// Flat layout: [((tap*nchunk + chunk)*ntiles + ct)*32 + lane][16 halves]
//   lane <16 : col = lane,      K = h<8 ? h : 8 + h          (0..7, 16..23)
//   lane>=16 : col = lane - 16, K = h<8 ? 8 + h : 16 + h     (8..15, 24..31)
//   cin = chunk*32 + K, cout = ct*16 + col; out-of-range -> 0.
// ---------------------------------------------------------------------------
__global__ __launch_bounds__(256) void pack_w_kernel(
    const float* __restrict__ w, _Float16* __restrict__ wp,
    int Cin, int Cout, int Cinp, int Coutp) {
  int nchunk = Cinp >> 5;
  int ntiles = Coutp >> 4;
  long long total = (long long)9 * nchunk * ntiles * 32 * 16;
  long long i = (long long)blockIdx.x * 256 + threadIdx.x;
  if (i >= total) return;
  int h    = (int)(i & 15);
  int lane = (int)((i >> 4) & 31);
  long long r = i >> 9;
  int ct    = (int)(r % ntiles); r /= ntiles;
  int chunk = (int)(r % nchunk);
  int tap   = (int)(r / nchunk);
  int col   = lane & 15;
  int kk    = ((h >> 3) << 4) + ((lane >> 4) << 3) + (h & 7);
  int cin   = chunk * 32 + kk;
  int cout  = ct * 16 + col;
  float v = 0.0f;
  if (cin < Cin && cout < Cout)
    v = w[((size_t)cout * Cin + cin) * 9 + tap];
  wp[i] = (_Float16)v;
}

// ---------------------------------------------------------------------------
// 3x3 SAME conv, implicit GEMM:
//   M = 16 pixels (per wave), N = 16 couts (2 tiles per wave), K = Cinp*9.
// 8 waves per block -> 128 px x 32 couts per workgroup.
// Epilogue: +bias, optional ReLU, optional skip add (relu(conv)+skip).
// ---------------------------------------------------------------------------
__device__ __forceinline__ void conv_store_tile(
    v8f acc, int ct, int tileBase, int lane, int P, int Coutp, int coutReal,
    int relu, const float* __restrict__ bias, const _Float16* __restrict__ skip,
    _Float16* __restrict__ out) {
  const int nCol  = lane & 15;
  const int mBase = (lane >> 4) * 8;
  const int co    = ct * 16 + nCol;
  const float bv  = (co < coutReal) ? bias[co] : 0.0f;
#pragma unroll
  for (int r = 0; r < 8; ++r) {
    const int p2 = tileBase + mBase + r;
    if (p2 < P) {
      float v = acc[r] + bv;
      if (relu) v = fmaxf(v, 0.0f);
      if (skip) v += (float)skip[(size_t)p2 * Coutp + co];
      out[(size_t)p2 * Coutp + co] = (_Float16)v;
    }
  }
}

__device__ __forceinline__ v16h load_a_frag(const _Float16* p) {
  const v8h* ap = (const v8h*)p;
  v8h lo = ap[0];   // K 0..7  (or 8..15 for hi lanes)
  v8h hi = ap[2];   // K 16..23 (or 24..31)
  return __builtin_shufflevector(lo, hi, 0, 1, 2, 3, 4, 5, 6, 7,
                                 8, 9, 10, 11, 12, 13, 14, 15);
}

template <int NCHUNK>
__global__ __launch_bounds__(256) void conv3x3_wmma_kernel(
    const _Float16* __restrict__ act,    // [H*W][Cinp] NHWC fp16
    const _Float16* __restrict__ wpack,  // packed B fragments
    const float*    __restrict__ bias,   // [coutReal] fp32
    const _Float16* __restrict__ skip,   // nullable [H*W][Coutp]
    _Float16*       __restrict__ out,    // [H*W][Coutp]
    int H, int W, int Coutp, int coutReal, int relu) {
  const int lane = threadIdx.x & 31;
  const int wave = threadIdx.x >> 5;
  const int P = H * W;
  const int Cinp = NCHUNK * 32;
  const int ntiles = Coutp >> 4;
  const int tileBase = (blockIdx.x * 8 + wave) * 16;
  const int ct0 = blockIdx.y * 2;
  const int ct1 = ct0 + 1;

  // A-matrix row handled by this lane (lanes L and L+16 share row M = L&15).
  const int mRow = lane & 15;
  const int pix  = tileBase + mRow;
  const int py   = pix / W;
  const int px   = pix - py * W;
  const int chOff = (lane >> 4) ? 8 : 0;  // fp16 offset inside the 32-chunk

  v8f acc0 = {};
  v8f acc1 = {};
  const v16h* wp = (const v16h*)wpack;

  for (int tap = 0; tap < 9; ++tap) {
    const int dy = tap / 3 - 1;
    const int dx = tap - (tap / 3) * 3 - 1;
    const int yy = py + dy;
    const int xx = px + dx;
    const bool valid = (pix < P) && (yy >= 0) && (yy < H) && (xx >= 0) && (xx < W);
    const size_t abase = valid ? ((size_t)(yy * W + xx) * Cinp + chOff) : 0;
    // Warm the WGP-level cache with the next tap's weight fragments
    // (global_prefetch_b8; weights for the whole net fit in L2).
    if (tap < 8) {
      __builtin_prefetch((const void*)(wp + ((size_t)((tap + 1) * NCHUNK) * ntiles + ct0) * 32), 0, 1);
    }
#pragma unroll
    for (int ch = 0; ch < NCHUNK; ++ch) {
      v16h a = {};
      if (valid) a = load_a_frag(act + abase + ch * 32);
      const size_t wrow = (size_t)(tap * NCHUNK + ch) * ntiles;
      v16h b0 = wp[(wrow + ct0) * 32 + lane];
      v16h b1 = wp[(wrow + ct1) * 32 + lane];
      acc0 = __builtin_amdgcn_wmma_f32_16x16x32_f16(false, a, false, b0,
                                                    (short)0, acc0, false, false);
      acc1 = __builtin_amdgcn_wmma_f32_16x16x32_f16(false, a, false, b1,
                                                    (short)0, acc1, false, false);
    }
  }

  conv_store_tile(acc0, ct0, tileBase, lane, P, Coutp, coutReal, relu, bias, skip, out);
  conv_store_tile(acc1, ct1, tileBase, lane, P, Coutp, coutReal, relu, bias, skip, out);
}

// Generic fallback (rolled chunk loop) for any other channel count.
__global__ __launch_bounds__(256) void conv3x3_wmma_generic(
    const _Float16* __restrict__ act, const _Float16* __restrict__ wpack,
    const float* __restrict__ bias, const _Float16* __restrict__ skip,
    _Float16* __restrict__ out,
    int H, int W, int Cinp, int Coutp, int coutReal, int relu) {
  const int lane = threadIdx.x & 31;
  const int wave = threadIdx.x >> 5;
  const int P = H * W;
  const int nchunk = Cinp >> 5;
  const int ntiles = Coutp >> 4;
  const int tileBase = (blockIdx.x * 8 + wave) * 16;
  const int ct0 = blockIdx.y * 2;
  const int ct1 = ct0 + 1;
  const int mRow = lane & 15;
  const int pix  = tileBase + mRow;
  const int py   = pix / W;
  const int px   = pix - py * W;
  const int chOff = (lane >> 4) ? 8 : 0;

  v8f acc0 = {};
  v8f acc1 = {};
  const v16h* wp = (const v16h*)wpack;

  for (int tap = 0; tap < 9; ++tap) {
    const int dy = tap / 3 - 1;
    const int dx = tap - (tap / 3) * 3 - 1;
    const int yy = py + dy;
    const int xx = px + dx;
    const bool valid = (pix < P) && (yy >= 0) && (yy < H) && (xx >= 0) && (xx < W);
    const size_t abase = valid ? ((size_t)(yy * W + xx) * Cinp + chOff) : 0;
    for (int ch = 0; ch < nchunk; ++ch) {
      v16h a = {};
      if (valid) a = load_a_frag(act + abase + ch * 32);
      const size_t wrow = (size_t)(tap * nchunk + ch) * ntiles;
      v16h b0 = wp[(wrow + ct0) * 32 + lane];
      v16h b1 = wp[(wrow + ct1) * 32 + lane];
      acc0 = __builtin_amdgcn_wmma_f32_16x16x32_f16(false, a, false, b0,
                                                    (short)0, acc0, false, false);
      acc1 = __builtin_amdgcn_wmma_f32_16x16x32_f16(false, a, false, b1,
                                                    (short)0, acc1, false, false);
    }
  }

  conv_store_tile(acc0, ct0, tileBase, lane, P, Coutp, coutReal, relu, bias, skip, out);
  conv_store_tile(acc1, ct1, tileBase, lane, P, Coutp, coutReal, relu, bias, skip, out);
}

// ---------------------------------------------------------------------------
// 2x2 average pool, NHWC fp16.
// ---------------------------------------------------------------------------
__global__ __launch_bounds__(256) void pool2_kernel(
    const _Float16* __restrict__ in, _Float16* __restrict__ out,
    int Ho, int Wo, int C) {
  long long i = (long long)blockIdx.x * 256 + threadIdx.x;
  long long total = (long long)Ho * Wo * C;
  if (i >= total) return;
  int c = (int)(i % C);
  long long p = i / C;
  int xo = (int)(p % Wo);
  int yo = (int)(p / Wo);
  int Wi = Wo * 2;
  size_t b  = ((size_t)(yo * 2) * Wi + xo * 2) * C + c;
  size_t rs = (size_t)Wi * C;
  float s = (float)in[b] + (float)in[b + C] + (float)in[b + rs] + (float)in[b + rs + C];
  out[i] = (_Float16)(0.25f * s);
}

// ---------------------------------------------------------------------------
// Bilinear 2x upsample (half-pixel centers, edge clamped), NHWC fp16.
// ---------------------------------------------------------------------------
__global__ __launch_bounds__(256) void up2_kernel(
    const _Float16* __restrict__ in, _Float16* __restrict__ out,
    int Hi, int Wi, int C) {
  int Ho = Hi * 2, Wo = Wi * 2;
  long long i = (long long)blockIdx.x * 256 + threadIdx.x;
  long long total = (long long)Ho * Wo * C;
  if (i >= total) return;
  int c = (int)(i % C);
  long long p = i / C;
  int xo = (int)(p % Wo);
  int yo = (int)(p / Wo);
  float sy = (yo + 0.5f) * 0.5f - 0.5f;
  float sx = (xo + 0.5f) * 0.5f - 0.5f;
  int y0 = (int)floorf(sy);
  int x0 = (int)floorf(sx);
  float fy = sy - (float)y0;
  float fx = sx - (float)x0;
  int y0c = iclampi(y0, 0, Hi - 1), y1c = iclampi(y0 + 1, 0, Hi - 1);
  int x0c = iclampi(x0, 0, Wi - 1), x1c = iclampi(x0 + 1, 0, Wi - 1);
  float a00 = (float)in[((size_t)y0c * Wi + x0c) * C + c];
  float a01 = (float)in[((size_t)y0c * Wi + x1c) * C + c];
  float a10 = (float)in[((size_t)y1c * Wi + x0c) * C + c];
  float a11 = (float)in[((size_t)y1c * Wi + x1c) * C + c];
  float v = (1.0f - fy) * ((1.0f - fx) * a00 + fx * a01) +
            fy          * ((1.0f - fx) * a10 + fx * a11);
  out[i] = (_Float16)v;
}

// ---------------------------------------------------------------------------
// Final separable 51x51 adaptive convolution.
// Block = one row h, 64 consecutive w.  LDS-stage kh/kv (both images),
// then each of 192 threads does one (channel, w) column:
//   acc = sum_d kv[d] * sum_j img[c][clamp(h+d-25)][clamp(w+j-25)] * kh[j]
// summed over both images, clipped to [0,1].  Output fp32 NCHW.
// ---------------------------------------------------------------------------
__global__ __launch_bounds__(256) void sepconv_kernel(
    const float* __restrict__ img1, const float* __restrict__ img2,
    const _Float16* __restrict__ k1v, const _Float16* __restrict__ k1h,
    const _Float16* __restrict__ k2v, const _Float16* __restrict__ k2h,
    float* __restrict__ out, int H, int W) {
  __shared__ float skh1[64][SEP_KS + 1];
  __shared__ float skv1[64][SEP_KS + 1];
  __shared__ float skh2[64][SEP_KS + 1];
  __shared__ float skv2[64][SEP_KS + 1];

  const int segs = W / 64;
  const int h  = blockIdx.x / segs;
  const int w0 = (blockIdx.x % segs) * 64;

  for (int idx = threadIdx.x; idx < 64 * SEP_KS; idx += 256) {
    int pixl = idx / SEP_KS;
    int j    = idx - pixl * SEP_KS;
    size_t g = ((size_t)(h * W + w0 + pixl)) * 64 + j;   // k maps: [P][64] fp16
    skh1[pixl][j] = (float)k1h[g];
    skv1[pixl][j] = (float)k1v[g];
    skh2[pixl][j] = (float)k2h[g];
    skv2[pixl][j] = (float)k2v[g];
  }
  __syncthreads();

  const int t = threadIdx.x;
  if (t < 192) {
    const int c  = t / 64;
    const int lw = t - c * 64;
    const int w  = w0 + lw;
    float acc = 0.0f;
    for (int img = 0; img < 2; ++img) {
      const float* base = (img ? img2 : img1) + (size_t)c * H * W;
      const float* kh = img ? &skh2[lw][0] : &skh1[lw][0];
      const float* kv = img ? &skv2[lw][0] : &skv1[lw][0];
      for (int d = 0; d < SEP_KS; ++d) {
        int yy = iclampi(h + d - SEP_PAD, 0, H - 1);
        const float* row = base + (size_t)yy * W;
        float rs = 0.0f;
        for (int j = 0; j < SEP_KS; ++j) {
          int xx = iclampi(w + j - SEP_PAD, 0, W - 1);
          rs = fmaf(row[xx], kh[j], rs);
        }
        acc = fmaf(rs, kv[d], acc);
      }
    }
    out[((size_t)c * H + h) * W + w] = fminf(fmaxf(acc, 0.0f), 1.0f);
  }
}

// ---------------------------------------------------------------------------
// Host orchestration.
// ---------------------------------------------------------------------------
namespace {

struct Layer {
  const float* b;
  _Float16* wp;
  int Cin, Cout, Cinp, Coutp;
};

static inline int pad32(int c) { return (c + 31) & ~31; }

}  // namespace

extern "C" void kernel_launch(void* const* d_in, const int* in_sizes, int n_in,
                              void* d_out, int out_size, void* d_ws, size_t ws_size,
                              hipStream_t stream) {
  (void)out_size; (void)ws_size;
  if (n_in < 96) return;

  const int H0 = 256, W0 = 384, P0 = H0 * W0;
  const int H1 = 128, W1 = 192, P1 = H1 * W1;
  const int H2 = 64,  W2 = 96,  P2 = H2 * W2;
  const int H3 = 32,  W3 = 48,  P3 = H3 * W3;
  const int H4 = 16,  W4 = 24,  P4 = H4 * W4;
  const int H5 = 8,   W5 = 12,  P5 = H5 * W5;

  // ---- parameter-leaf index mapping --------------------------------------
  // Two plausible flattenings of the nested params dict:
  //  * jax tree-flatten (keys sorted):  conv1..conv9, h1, h2, up6..up9, v1, v2
  //  * python insertion order:          conv1..6, up6, conv7, up7, ..., v1,h1,v2,h2
  // Disambiguate from in_sizes[38]: sorted -> conv7.w0 (256*512*9 = 1179648),
  // insertion -> up6.w (512*512*9 = 2359296).
  int cw[9][3], cb[9][3], upw[4], upb[4], sw[4][4], sb[4][4];  // sub: 0=v1,1=h1,2=v2,3=h2
  const bool insertionOrder = (in_sizes[38] == 512 * 512 * 9);
  if (insertionOrder) {
    int p = 2;
    for (int k = 0; k < 6; ++k)
      for (int i = 0; i < 3; ++i) { cw[k][i] = p; cb[k][i] = p + 1; p += 2; }
    upw[0] = p; upb[0] = p + 1; p += 2;                       // up6
    for (int i = 0; i < 3; ++i) { cw[6][i] = p; cb[6][i] = p + 1; p += 2; }
    upw[1] = p; upb[1] = p + 1; p += 2;                       // up7
    for (int i = 0; i < 3; ++i) { cw[7][i] = p; cb[7][i] = p + 1; p += 2; }
    upw[2] = p; upb[2] = p + 1; p += 2;                       // up8
    for (int i = 0; i < 3; ++i) { cw[8][i] = p; cb[8][i] = p + 1; p += 2; }
    upw[3] = p; upb[3] = p + 1; p += 2;                       // up9
    for (int s = 0; s < 4; ++s)                               // v1,h1,v2,h2
      for (int i = 0; i < 4; ++i) { sw[s][i] = p; sb[s][i] = p + 1; p += 2; }
  } else {
    for (int k = 0; k < 9; ++k)
      for (int i = 0; i < 3; ++i) { cw[k][i] = 2 + k * 6 + i * 2; cb[k][i] = cw[k][i] + 1; }
    for (int i = 0; i < 4; ++i) { sw[1][i] = 56 + i * 2; sb[1][i] = sw[1][i] + 1; }  // h1
    for (int i = 0; i < 4; ++i) { sw[3][i] = 64 + i * 2; sb[3][i] = sw[3][i] + 1; }  // h2
    for (int j = 0; j < 4; ++j) { upw[j] = 72 + j * 2; upb[j] = upw[j] + 1; }        // up6..up9
    for (int i = 0; i < 4; ++i) { sw[0][i] = 80 + i * 2; sb[0][i] = sw[0][i] + 1; }  // v1
    for (int i = 0; i < 4; ++i) { sw[2][i] = 88 + i * 2; sb[2][i] = sw[2][i] + 1; }  // v2
  }

  // ---- deterministic bump allocator over d_ws ----------------------------
  uintptr_t base = (uintptr_t)d_ws;
  size_t off = 0;
  auto alloc = [&](size_t bytes) -> void* {
    off = (off + 255) & ~(size_t)255;
    void* p = (void*)(base + off);
    off += bytes;
    return p;
  };
  auto allocH = [&](size_t nhalf) -> _Float16* {
    return (_Float16*)alloc(nhalf * sizeof(_Float16));
  };

  // ---- pack weights into WMMA B-fragment layout --------------------------
  auto packLayer = [&](int wi, int bi, int Cin, int Cout) -> Layer {
    Layer L;
    L.Cin = Cin; L.Cout = Cout;
    L.Cinp = pad32(Cin); L.Coutp = pad32(Cout);
    size_t n = (size_t)9 * (L.Cinp / 32) * (L.Coutp / 16) * 32 * 16;
    L.wp = allocH(n);
    L.b = (const float*)d_in[bi];
    int blocks = (int)((n + 255) / 256);
    pack_w_kernel<<<blocks, 256, 0, stream>>>((const float*)d_in[wi], L.wp,
                                              Cin, Cout, L.Cinp, L.Coutp);
    return L;
  };

  Layer Lc[9][3], Lup[4], Lsub[4][4];
  const int tch[9][2] = {{6, 32},   {32, 64},  {64, 128}, {128, 256}, {256, 512},
                         {512, 512}, {512, 256}, {256, 128}, {128, 64}};
  for (int k = 0; k < 9; ++k)
    for (int i = 0; i < 3; ++i)
      Lc[k][i] = packLayer(cw[k][i], cb[k][i], i == 0 ? tch[k][0] : tch[k][1], tch[k][1]);
  const int uch[4] = {512, 256, 128, 64};
  for (int j = 0; j < 4; ++j) Lup[j] = packLayer(upw[j], upb[j], uch[j], uch[j]);
  const int sch[4][2] = {{64, 64}, {64, 64}, {64, 51}, {51, 51}};
  for (int s = 0; s < 4; ++s)
    for (int i = 0; i < 4; ++i)
      Lsub[s][i] = packLayer(sw[s][i], sb[s][i], sch[i][0], sch[i][1]);

  // ---- activation buffers (fp16 NHWC, padded channels) -------------------
  _Float16* x0  = allocH((size_t)P0 * 32);
  _Float16* L0a = allocH((size_t)P0 * 64);
  _Float16* L0b = allocH((size_t)P0 * 64);
  _Float16* kmap[4];
  for (int s = 0; s < 4; ++s) kmap[s] = allocH((size_t)P0 * 64);
  _Float16* L1a = allocH((size_t)P1 * 64);
  _Float16* L1b = allocH((size_t)P1 * 64);
  _Float16* c2  = allocH((size_t)P1 * 64);
  _Float16* u9  = allocH((size_t)P1 * 64);
  _Float16* L2a = allocH((size_t)P2 * 128);
  _Float16* L2b = allocH((size_t)P2 * 128);
  _Float16* c3  = allocH((size_t)P2 * 128);
  _Float16* L3a = allocH((size_t)P3 * 256);
  _Float16* L3b = allocH((size_t)P3 * 256);
  _Float16* c4  = allocH((size_t)P3 * 256);
  _Float16* L4a = allocH((size_t)P4 * 512);
  _Float16* L4b = allocH((size_t)P4 * 512);
  _Float16* c5  = allocH((size_t)P4 * 512);
  _Float16* L5a = allocH((size_t)P5 * 512);
  _Float16* L5b = allocH((size_t)P5 * 512);

  // ---- launch helpers ----------------------------------------------------
  auto conv = [&](const _Float16* in, _Float16* out, const Layer& L, int H, int W,
                  bool relu, const _Float16* skip) {
    dim3 g((unsigned)((H * W + 127) / 128), (unsigned)(L.Coutp / 32), 1);
    const int rl = relu ? 1 : 0;
    switch (L.Cinp >> 5) {
      case 1:
        conv3x3_wmma_kernel<1><<<g, 256, 0, stream>>>(in, L.wp, L.b, skip, out,
                                                      H, W, L.Coutp, L.Cout, rl);
        break;
      case 2:
        conv3x3_wmma_kernel<2><<<g, 256, 0, stream>>>(in, L.wp, L.b, skip, out,
                                                      H, W, L.Coutp, L.Cout, rl);
        break;
      case 4:
        conv3x3_wmma_kernel<4><<<g, 256, 0, stream>>>(in, L.wp, L.b, skip, out,
                                                      H, W, L.Coutp, L.Cout, rl);
        break;
      case 8:
        conv3x3_wmma_kernel<8><<<g, 256, 0, stream>>>(in, L.wp, L.b, skip, out,
                                                      H, W, L.Coutp, L.Cout, rl);
        break;
      case 16:
        conv3x3_wmma_kernel<16><<<g, 256, 0, stream>>>(in, L.wp, L.b, skip, out,
                                                       H, W, L.Coutp, L.Cout, rl);
        break;
      default:
        conv3x3_wmma_generic<<<g, 256, 0, stream>>>(in, L.wp, L.b, skip, out,
                                                    H, W, L.Cinp, L.Coutp, L.Cout, rl);
        break;
    }
  };
  auto pool = [&](const _Float16* in, _Float16* out, int Ho, int Wo, int C) {
    long long n = (long long)Ho * Wo * C;
    pool2_kernel<<<(unsigned)((n + 255) / 256), 256, 0, stream>>>(in, out, Ho, Wo, C);
  };
  auto up2 = [&](const _Float16* in, _Float16* out, int Hi, int Wi, int C) {
    long long n = (long long)Hi * 2 * Wi * 2 * C;
    up2_kernel<<<(unsigned)((n + 255) / 256), 256, 0, stream>>>(in, out, Hi, Wi, C);
  };

  const float* img1 = (const float*)d_in[0];
  const float* img2 = (const float*)d_in[1];

  // ---- forward -----------------------------------------------------------
  pack_input_kernel<<<(P0 * 32 + 255) / 256, 256, 0, stream>>>(img1, img2, x0, P0);

  // encoder
  conv(x0,  L0a, Lc[0][0], H0, W0, true, nullptr);
  conv(L0a, L0b, Lc[0][1], H0, W0, true, nullptr);
  conv(L0b, L0a, Lc[0][2], H0, W0, true, nullptr);        // c1
  pool(L0a, L1a, H1, W1, 32);                              // p1
  conv(L1a, L1b, Lc[1][0], H1, W1, true, nullptr);
  conv(L1b, L1a, Lc[1][1], H1, W1, true, nullptr);
  conv(L1a, c2,  Lc[1][2], H1, W1, true, nullptr);         // c2 (skip)
  pool(c2, L2a, H2, W2, 64);                               // p2
  conv(L2a, L2b, Lc[2][0], H2, W2, true, nullptr);
  conv(L2b, L2a, Lc[2][1], H2, W2, true, nullptr);
  conv(L2a, c3,  Lc[2][2], H2, W2, true, nullptr);         // c3 (skip)
  pool(c3, L3a, H3, W3, 128);                              // p3
  conv(L3a, L3b, Lc[3][0], H3, W3, true, nullptr);
  conv(L3b, L3a, Lc[3][1], H3, W3, true, nullptr);
  conv(L3a, c4,  Lc[3][2], H3, W3, true, nullptr);         // c4 (skip)
  pool(c4, L4a, H4, W4, 256);                              // p4
  conv(L4a, L4b, Lc[4][0], H4, W4, true, nullptr);
  conv(L4b, L4a, Lc[4][1], H4, W4, true, nullptr);
  conv(L4a, c5,  Lc[4][2], H4, W4, true, nullptr);         // c5 (skip)
  pool(c5, L5a, H5, W5, 512);                              // p5
  conv(L5a, L5b, Lc[5][0], H5, W5, true, nullptr);
  conv(L5b, L5a, Lc[5][1], H5, W5, true, nullptr);
  conv(L5a, L5b, Lc[5][2], H5, W5, true, nullptr);         // c6

  // decoder
  up2(L5b, L4a, H5, W5, 512);
  conv(L4a, L4b, Lup[0], H4, W4, true, c5);                // u6 = relu(conv)+c5
  conv(L4b, L4a, Lc[6][0], H4, W4, true, nullptr);
  conv(L4a, L4b, Lc[6][1], H4, W4, true, nullptr);
  conv(L4b, L4a, Lc[6][2], H4, W4, true, nullptr);         // c7
  up2(L4a, L3a, H4, W4, 256);
  conv(L3a, L3b, Lup[1], H3, W3, true, c4);                // u7
  conv(L3b, L3a, Lc[7][0], H3, W3, true, nullptr);
  conv(L3a, L3b, Lc[7][1], H3, W3, true, nullptr);
  conv(L3b, L3a, Lc[7][2], H3, W3, true, nullptr);         // c8
  up2(L3a, L2a, H3, W3, 128);
  conv(L2a, L2b, Lup[2], H2, W2, true, c3);                // u8
  conv(L2b, L2a, Lc[8][0], H2, W2, true, nullptr);
  conv(L2a, L2b, Lc[8][1], H2, W2, true, nullptr);
  conv(L2b, L2a, Lc[8][2], H2, W2, true, nullptr);         // c9
  up2(L2a, L1a, H2, W2, 64);
  conv(L1a, u9, Lup[3], H1, W1, true, c2);                 // u9

  // four kernel-prediction subnets (0=v1, 1=h1, 2=v2, 3=h2)
  for (int s = 0; s < 4; ++s) {
    conv(u9,  L1a, Lsub[s][0], H1, W1, true, nullptr);
    conv(L1a, L1b, Lsub[s][1], H1, W1, true, nullptr);
    conv(L1b, L1a, Lsub[s][2], H1, W1, true, nullptr);     // 51ch (pad 64)
    up2(L1a, L0a, H1, W1, 64);
    conv(L0a, kmap[s], Lsub[s][3], H0, W0, false, nullptr);
  }

  // final separable 51x51 adaptive convolution -> fp32 NCHW output
  sepconv_kernel<<<H0 * (W0 / 64), 256, 0, stream>>>(
      img1, img2, kmap[0], kmap[1], kmap[2], kmap[3], (float*)d_out, H0, W0);
}
